// StageBlock_20160576487504
// MI455X (gfx1250) — matrix operand order repaired and checked by reading, not compiled, not run on hardware
//
#include <hip/hip_runtime.h>

// Problem constants (match reference)
#define BB   8
#define NN   4096
#define SS   512
#define KK   32
#define DIN  64
#define DD   128      // channel dim entering Double_SA
#define DOUT 128
#define EPSF 1e-5f

typedef __attribute__((ext_vector_type(16))) _Float16 v16h;
typedef __attribute__((ext_vector_type(8)))  float    v8f;
typedef __attribute__((ext_vector_type(2)))  _Float16 h2;
typedef __attribute__((ext_vector_type(8)))  _Float16 h8;   // 16 bytes
typedef int i4 __attribute__((vector_size(16)));            // 4 x i32 = 128 bits

// ---------------------------------------------------------------------------
// Async global -> LDS 16B copy (CDNA5 GLOBAL_LOAD_ASYNC_TO_LDS_B128) with
// portable fallback (global_load_b128 + ds_store_b128).
// Builtin signature (probed): (i4 AS1*, i4 AS3*, imm offset, imm cpol)
// ---------------------------------------------------------------------------
#if defined(__HIP_DEVICE_COMPILE__) && __has_builtin(__builtin_amdgcn_global_load_async_to_lds_b128)
#define HAVE_ASYNC_LDS 1
#else
#define HAVE_ASYNC_LDS 0
#endif

__device__ __forceinline__ void copy16_g2l(const _Float16* g, _Float16* l) {
#if HAVE_ASYNC_LDS
  __builtin_amdgcn_global_load_async_to_lds_b128(
      (__attribute__((address_space(1))) i4*)(g),
      (__attribute__((address_space(3))) i4*)(l), 0, 0);
#else
  *(h8*)l = *(const h8*)g;
#endif
}

__device__ __forceinline__ void wait_async_copies() {
#if HAVE_ASYNC_LDS
#if __has_builtin(__builtin_amdgcn_s_wait_asynccnt)
  __builtin_amdgcn_s_wait_asynccnt(0);
#else
  asm volatile("s_wait_asynccnt 0x0" ::: "memory");
#endif
#endif
}

// ---------------------------------------------------------------------------
// 1. Farthest point sampling: one block per batch, LDS-resident distance array
// ---------------------------------------------------------------------------
__global__ void fps_kernel(const float* __restrict__ xyz, const int* __restrict__ start,
                           int* __restrict__ fps_idx, float* __restrict__ center_ws,
                           float* __restrict__ center_out) {
  const int b = blockIdx.x;
  const int t = threadIdx.x;                     // 256 threads
  __shared__ float dist[NN];
  __shared__ float rmax[256];
  __shared__ int   rarg[256];
  __shared__ int   s_far;
  const float* X = xyz + (long long)b * NN * 3;
  for (int i = t; i < NN; i += 256) dist[i] = 1e10f;
  if (t == 0) s_far = start[b];
  __syncthreads();
  for (int it = 0; it < SS; ++it) {
    const int far = s_far;
    const float cx = X[far * 3 + 0], cy = X[far * 3 + 1], cz = X[far * 3 + 2];
    if (t == 0) {
      fps_idx[b * SS + it] = far;
      float* cw = center_ws + (b * SS + it) * 3;
      cw[0] = cx; cw[1] = cy; cw[2] = cz;
      float* co = center_out + (b * SS + it) * 3;
      co[0] = cx; co[1] = cy; co[2] = cz;
    }
    float lbest = -1.0f; int larg = 0;
    for (int i = t; i < NN; i += 256) {
      const float dx = X[i * 3 + 0] - cx;
      const float dy = X[i * 3 + 1] - cy;
      const float dz = X[i * 3 + 2] - cz;
      float d = fminf(dist[i], dx * dx + dy * dy + dz * dz);
      dist[i] = d;
      if (d > lbest) { lbest = d; larg = i; }
    }
    rmax[t] = lbest; rarg[t] = larg;
    __syncthreads();
    for (int s2 = 128; s2 > 0; s2 >>= 1) {
      if (t < s2) {
        const float ov = rmax[t + s2]; const int oi = rarg[t + s2];
        if (ov > rmax[t] || (ov == rmax[t] && oi < rarg[t])) { rmax[t] = ov; rarg[t] = oi; }
      }
      __syncthreads();
    }
    if (t == 0) s_far = rarg[0];
    __syncthreads();
  }
}

// ---------------------------------------------------------------------------
// 2. Brute-force KNN: one thread per center, insertion-sorted top-32 (ascending)
// ---------------------------------------------------------------------------
__global__ void knn_kernel(const float* __restrict__ xyz, const float* __restrict__ center,
                           int* __restrict__ knn_idx, float* __restrict__ knn_dist) {
  const int g = blockIdx.x * blockDim.x + threadIdx.x;   // b*SS + s
  if (g >= BB * SS) return;
  const int b = g / SS;
  const float* X = xyz + (long long)b * NN * 3;
  const float cx = center[g * 3 + 0], cy = center[g * 3 + 1], cz = center[g * 3 + 2];
  float bd[KK]; int bi[KK];
  #pragma unroll
  for (int j = 0; j < KK; ++j) { bd[j] = 3.4e38f; bi[j] = 0; }
  for (int n = 0; n < NN; ++n) {
    const float dx = X[n * 3 + 0] - cx;
    const float dy = X[n * 3 + 1] - cy;
    const float dz = X[n * 3 + 2] - cz;
    const float d = dx * dx + dy * dy + dz * dz;
    if (d < bd[KK - 1]) {
      int j = KK - 1;
      while (j > 0 && bd[j - 1] > d) { bd[j] = bd[j - 1]; bi[j] = bi[j - 1]; --j; }
      bd[j] = d; bi[j] = n;
    }
  }
  for (int j = 0; j < KK; ++j) { knn_idx[g * KK + j] = bi[j]; knn_dist[g * KK + j] = bd[j]; }
}

// ---------------------------------------------------------------------------
// 3. Gather + positional-embedding MLP (10->32->64) + concat -> x (f16 rows)
//    x row = [pos(64) | grouped_pts(64)]; x is also GAM's "flat" [B, S, K*D]
// ---------------------------------------------------------------------------
__global__ void group_pos_kernel(const float* __restrict__ xyz, const float* __restrict__ points,
                                 const float* __restrict__ center,
                                 const int* __restrict__ knn_idx, const float* __restrict__ knn_dist,
                                 const float* __restrict__ w1, const float* __restrict__ b1,
                                 const float* __restrict__ bn1,
                                 const float* __restrict__ w2, const float* __restrict__ b2,
                                 const float* __restrict__ bn2,
                                 _Float16* __restrict__ x16) {
  const int r = blockIdx.x * blockDim.x + threadIdx.x;   // row in [0, B*S*K)
  if (r >= BB * SS * KK) return;
  const int g = r / KK;
  const int b = g / SS;
  const int n = knn_idx[r];
  const float* p = xyz + ((long long)b * NN + n) * 3;
  const float* c = center + (long long)g * 3;
  float feat[10];
  feat[0] = c[0]; feat[1] = c[1]; feat[2] = c[2];
  feat[3] = p[0]; feat[4] = p[1]; feat[5] = p[2];
  feat[6] = p[0] - c[0]; feat[7] = p[1] - c[1]; feat[8] = p[2] - c[2];
  feat[9] = knn_dist[r];
  float h[32];
  #pragma unroll
  for (int o = 0; o < 32; ++o) {
    float a = b1[o];
    #pragma unroll
    for (int j = 0; j < 10; ++j) a += feat[j] * w1[j * 32 + o];
    a = bn1[o] * (a - bn1[64 + o]) * rsqrtf(bn1[96 + o] + EPSF) + bn1[32 + o];
    h[o] = fmaxf(a, 0.0f);
  }
  _Float16* xr = x16 + (long long)r * DD;
  for (int o = 0; o < 64; ++o) {
    float a = b2[o];
    #pragma unroll
    for (int j = 0; j < 32; ++j) a += h[j] * w2[j * 64 + o];
    a = bn2[o] * (a - bn2[128 + o]) * rsqrtf(bn2[192 + o] + EPSF) + bn2[64 + o];
    xr[o] = (_Float16)fmaxf(a, 0.0f);
  }
  const float* pt = points + ((long long)b * NN + n) * DIN;
  for (int o = 0; o < 64; ++o) xr[64 + o] = (_Float16)pt[o];
}

// ---------------------------------------------------------------------------
// 4a. f32 -> f16 weight conversion
// ---------------------------------------------------------------------------
__global__ void f2h_kernel(const float* __restrict__ src, _Float16* __restrict__ dst, int n) {
  const int i = blockIdx.x * blockDim.x + threadIdx.x;
  if (i < n) dst[i] = (_Float16)src[i];
}

// 4b. Pack wq|wk -> [128,32] f16 and bq|bk -> [32] f32 (fused QK projection)
__global__ void qk_pack_kernel(const float* __restrict__ wq, const float* __restrict__ wk,
                               const float* __restrict__ bq, const float* __restrict__ bk,
                               _Float16* __restrict__ wqk, float* __restrict__ bqk) {
  const int i = blockIdx.x * blockDim.x + threadIdx.x;
  if (i < DD * 32) {
    const int k = i >> 5, n = i & 31;
    wqk[i] = (_Float16)(n < 16 ? wq[k * 16 + n] : wk[k * 16 + (n - 16)]);
  }
  if (i < 32) bqk[i] = (i < 16) ? bq[i] : bk[i - 16];
}

// ---------------------------------------------------------------------------
// 5. Tiled batched WMMA GEMM: C[M,N] = A[M,Kd] * B[Kd,N] (+bias)(+BN)(+ReLU)
//    Block = WGM*WGN waves; block tile (WGM*WTM*16) x (WGN*WTN*16); K-step 32.
//    A and B tiles staged in LDS (async global->LDS when available); B stored
//    N-major in LDS so fragments are packed-h2 ds_load_b32 reads.
//    A/B fragments use contiguous k order, k = 16*(lane/16) + s, in both operands (wave32).
// ---------------------------------------------------------------------------
template <int WGM, int WGN, int WTM, int WTN, bool BT>
__global__ void gemm_tile_kernel(const _Float16* __restrict__ A, const _Float16* __restrict__ Bm,
                                 const float* __restrict__ bias, const float* __restrict__ bnp, int bnC,
                                 float* __restrict__ Cf, _Float16* __restrict__ Ch,
                                 int Kd, int lda, int ldb, int ldc,
                                 long long sA, long long sB, long long sC, int relu) {
  constexpr int BM = WGM * WTM * 16;
  constexpr int BN = WGN * WTN * 16;
  constexpr int THREADS = WGM * WGN * 32;
  constexpr int KS = 32;
  __shared__ _Float16 As[BM * KS];
  __shared__ _Float16 Bs[BN * KS];          // N-major: Bs[n*KS + k]
  const int tid = threadIdx.x;
  const int lane = tid & 31;
  const int wid = tid >> 5;
  const int waveM = wid % WGM, waveN = wid / WGM;
  const int n0 = blockIdx.x * BN;
  const int m0 = blockIdx.y * BM;
  A  += (long long)blockIdx.z * sA;
  Bm += (long long)blockIdx.z * sB;
  const long long cOff = (long long)blockIdx.z * sC;
  const int ml = lane & 15, hi = lane >> 4;

  v8f acc[WTM][WTN];
  #pragma unroll
  for (int i = 0; i < WTM; ++i)
    #pragma unroll
    for (int j = 0; j < WTN; ++j)
      #pragma unroll
      for (int e = 0; e < 8; ++e) acc[i][j][e] = 0.0f;

  for (int k0 = 0; k0 < Kd; k0 += KS) {
    // ---- stage A tile: BM x 32 halves, 16B segments, coalesced ----
    #pragma unroll
    for (int s = tid; s < BM * 4; s += THREADS) {
      const int row = s >> 2, seg = s & 3;
      copy16_g2l(A + (long long)(m0 + row) * lda + k0 + seg * 8, &As[row * KS + seg * 8]);
    }
    // ---- stage B tile ----
    if (BT) {
      // B is N-major in memory: rows are N, K contiguous -> same pattern as A
      #pragma unroll
      for (int s = tid; s < BN * 4; s += THREADS) {
        const int row = s >> 2, seg = s & 3;
        copy16_g2l(Bm + (long long)(n0 + row) * ldb + k0 + seg * 8, &Bs[row * KS + seg * 8]);
      }
    } else {
      // B is K-major: read 16B runs along N, transpose-scatter into LDS
      for (int s = tid; s < KS * (BN / 8); s += THREADS) {
        const int k = s / (BN / 8), nb = s % (BN / 8);
        const h8 v = *(const h8*)(Bm + (long long)(k0 + k) * ldb + n0 + nb * 8);
        #pragma unroll
        for (int j = 0; j < 8; ++j) Bs[(nb * 8 + j) * KS + k] = v[j];
      }
    }
    wait_async_copies();
    __syncthreads();

    // ---- fragments from LDS (packed h2 -> ds_load_b32) ----
    v16h af[WTM], bfv[WTN];
    #pragma unroll
    for (int tm = 0; tm < WTM; ++tm) {
      const _Float16* ar = &As[(waveM * WTM * 16 + tm * 16 + ml) * KS + hi * 16];
      #pragma unroll
      for (int v = 0; v < 8; ++v) {
        const int off = 2 * v;
        const h2 pr = *(const h2*)(ar + off);
        af[tm][2 * v] = pr[0]; af[tm][2 * v + 1] = pr[1];
      }
    }
    #pragma unroll
    for (int tn = 0; tn < WTN; ++tn) {
      const _Float16* br = &Bs[(waveN * WTN * 16 + tn * 16 + ml) * KS + hi * 16];
      #pragma unroll
      for (int j = 0; j < 8; ++j) {
        const h2 pr = *(const h2*)(br + 2 * j);
        bfv[tn][2 * j] = pr[0]; bfv[tn][2 * j + 1] = pr[1];
      }
    }
    #pragma unroll
    for (int tm = 0; tm < WTM; ++tm)
      #pragma unroll
      for (int tn = 0; tn < WTN; ++tn)
        acc[tm][tn] = __builtin_amdgcn_wmma_f32_16x16x32_f16(
            false, af[tm], false, bfv[tn], (short)0, acc[tm][tn], false, false);
    __syncthreads();
  }

  // ---- epilogue: bias + BN + ReLU, f32 or f16 store ----
  #pragma unroll
  for (int tm = 0; tm < WTM; ++tm) {
    #pragma unroll
    for (int tn = 0; tn < WTN; ++tn) {
      #pragma unroll
      for (int r = 0; r < 8; ++r) {
        const int m = m0 + waveM * WTM * 16 + tm * 16 + r + 8 * hi;
        const int n = n0 + waveN * WTN * 16 + tn * 16 + ml;
        float val = acc[tm][tn][r];
        if (bias) val += bias[n];
        if (bnp) val = bnp[n] * (val - bnp[2 * bnC + n]) * rsqrtf(bnp[3 * bnC + n] + EPSF) + bnp[bnC + n];
        if (relu) val = fmaxf(val, 0.0f);
        const long long idx = cOff + (long long)m * ldc + n;
        if (Cf) Cf[idx] = val; else Ch[idx] = (_Float16)val;
      }
    }
  }
}

// ---------------------------------------------------------------------------
// 6. Neighbor self-attention core: per-group 32x32 softmax(QK^T) @ V, residual.
//    QK fused layout: row r -> q = QK[r*32 + 0..15], k = QK[r*32 + 16..31].
//    Writes nei into dsa columns [0,128).  Block = 128 threads, one group/block.
// ---------------------------------------------------------------------------
__global__ void nsa_kernel(const float* __restrict__ QK,
                           const _Float16* __restrict__ V, const _Float16* __restrict__ x16,
                           const float* __restrict__ gamma, _Float16* __restrict__ dsa) {
  const int g = blockIdx.x;          // b*SS + s
  const int t = threadIdx.x;         // 0..127
  __shared__ float sq[32][16], sk[32][16], sat[32][32];
  for (int i = t; i < 32 * 32; i += 128) {
    const int row = i >> 5, col = i & 31;
    const float v = QK[((long long)g * 32 + row) * 32 + col];
    if (col < 16) sq[row][col] = v; else sk[row][col - 16] = v;
  }
  __syncthreads();
  for (int e = t; e < 1024; e += 128) {
    const int i = e >> 5, j = e & 31;
    float a = 0.0f;
    #pragma unroll
    for (int c2 = 0; c2 < 16; ++c2) a += sq[i][c2] * sk[j][c2];
    sat[i][j] = a;
  }
  __syncthreads();
  if (t < 32) {
    float mx = -3.4e38f;
    for (int j = 0; j < 32; ++j) mx = fmaxf(mx, sat[t][j]);
    float sm = 0.0f;
    for (int j = 0; j < 32; ++j) { const float e2 = __expf(sat[t][j] - mx); sat[t][j] = e2; sm += e2; }
    const float inv = 1.0f / sm;
    for (int j = 0; j < 32; ++j) sat[t][j] *= inv;
  }
  __syncthreads();
  const int row = t >> 2, d0 = (t & 3) * 32;
  float acc[32];
  #pragma unroll
  for (int d = 0; d < 32; ++d) acc[d] = 0.0f;
  for (int j = 0; j < 32; ++j) {
    const float w = sat[row][j];
    const _Float16* vr = V + ((long long)g * 32 + j) * DD + d0;
    #pragma unroll
    for (int d = 0; d < 32; ++d) acc[d] += w * (float)vr[d];
  }
  const float gm = gamma[0];
  const _Float16* xr = x16 + ((long long)g * 32 + row) * DD + d0;
  _Float16* o = dsa + ((long long)g * 32 + row) * 256 + d0;
  for (int d = 0; d < 32; ++d) o[d] = (_Float16)(gm * acc[d] + (float)xr[d]);
}

// ---------------------------------------------------------------------------
// 7. GAM row softmax over 512 logits -> f16 attention weights
// ---------------------------------------------------------------------------
__global__ void gam_softmax_kernel(const float* __restrict__ scores, _Float16* __restrict__ ga) {
  const int row = blockIdx.x;        // b*SS + s
  const int t = threadIdx.x;         // 256
  const float* sr = scores + (long long)row * SS;
  __shared__ float red[256];
  float lm = -3.4e38f;
  for (int j = t; j < SS; j += 256) lm = fmaxf(lm, sr[j]);
  red[t] = lm; __syncthreads();
  for (int s2 = 128; s2 > 0; s2 >>= 1) { if (t < s2) red[t] = fmaxf(red[t], red[t + s2]); __syncthreads(); }
  const float mx = red[0]; __syncthreads();
  float ls = 0.0f;
  for (int j = t; j < SS; j += 256) ls += __expf(sr[j] - mx);
  red[t] = ls; __syncthreads();
  for (int s2 = 128; s2 > 0; s2 >>= 1) { if (t < s2) red[t] += red[t + s2]; __syncthreads(); }
  const float inv = 1.0f / red[0];
  _Float16* gr = ga + (long long)row * SS;
  for (int j = t; j < SS; j += 256) gr[j] = (_Float16)(__expf(sr[j] - mx) * inv);
}

// ---------------------------------------------------------------------------
// 8. grp = gam_gamma * (ga@flat) + x  -> dsa columns [128,256)
// ---------------------------------------------------------------------------
__global__ void grp_kernel(const _Float16* __restrict__ out2, const _Float16* __restrict__ x16,
                           const float* __restrict__ gamma, _Float16* __restrict__ dsa) {
  const long long i = (long long)blockIdx.x * blockDim.x + threadIdx.x;
  if (i >= (long long)BB * SS * KK * DD) return;
  const long long row = i / DD;
  const int d = (int)(i % DD);
  dsa[row * 256 + 128 + d] = (_Float16)(gamma[0] * (float)out2[i] + (float)x16[i]);
}

// ---------------------------------------------------------------------------
// 9. new_points = max_k relu(main + sc)
// ---------------------------------------------------------------------------
__global__ void final_kernel(const float* __restrict__ mainb, const float* __restrict__ scb,
                             float* __restrict__ out) {
  const int i = blockIdx.x * blockDim.x + threadIdx.x;   // (b*SS+s)*DOUT + d
  if (i >= BB * SS * DOUT) return;
  const int g = i / DOUT, d = i % DOUT;
  float mx = 0.0f;                                       // relu output >= 0
  for (int k = 0; k < KK; ++k) {
    const long long r = (long long)g * KK + k;
    mx = fmaxf(mx, fmaxf(mainb[r * DOUT + d] + scb[r * DOUT + d], 0.0f));
  }
  out[BB * SS * 3 + (long long)g * DOUT + d] = mx;
}

// ---------------------------------------------------------------------------
// Host launcher
// ---------------------------------------------------------------------------
static inline void* wsp(void* ws, size_t& off, size_t bytes) {
  void* p = (char*)ws + off;
  off += (bytes + 255) & ~(size_t)255;
  return p;
}

extern "C" void kernel_launch(void* const* d_in, const int* in_sizes, int n_in,
                              void* d_out, int out_size, void* d_ws, size_t ws_size,
                              hipStream_t stream) {
  (void)in_sizes; (void)n_in; (void)out_size; (void)ws_size;
  const float* xyz      = (const float*)d_in[0];
  const float* points   = (const float*)d_in[1];
  const int*   fpsstart = (const int*)  d_in[2];
  const float* pos_w1   = (const float*)d_in[3];
  const float* pos_b1   = (const float*)d_in[4];
  const float* pos_bn1  = (const float*)d_in[5];
  const float* pos_w2   = (const float*)d_in[6];
  const float* pos_b2   = (const float*)d_in[7];
  const float* pos_bn2  = (const float*)d_in[8];
  const float* nsa_wq   = (const float*)d_in[9];
  const float* nsa_bq   = (const float*)d_in[10];
  const float* nsa_wk   = (const float*)d_in[11];
  const float* nsa_bk   = (const float*)d_in[12];
  const float* nsa_wv   = (const float*)d_in[13];
  const float* nsa_bv   = (const float*)d_in[14];
  const float* nsa_gam  = (const float*)d_in[15];
  const float* gam_gam  = (const float*)d_in[16];
  const float* cat_w1   = (const float*)d_in[17];
  const float* cat_b1   = (const float*)d_in[18];
  const float* cat_bn1  = (const float*)d_in[19];
  const float* cat_w2   = (const float*)d_in[20];
  const float* cat_b2   = (const float*)d_in[21];
  const float* cat_bn2  = (const float*)d_in[22];
  const float* sc_w     = (const float*)d_in[23];
  const float* sc_b     = (const float*)d_in[24];
  const float* sc_bn    = (const float*)d_in[25];
  float* out = (float*)d_out;

  const long long R = (long long)BB * SS * KK;    // 131072 rows

  size_t off = 0;
  int*      fps_idx  = (int*)     wsp(d_ws, off, (size_t)BB * SS * 4);
  float*    center   = (float*)   wsp(d_ws, off, (size_t)BB * SS * 3 * 4);
  int*      knn_idx  = (int*)     wsp(d_ws, off, (size_t)R * 4);
  float*    knn_dist = (float*)   wsp(d_ws, off, (size_t)R * 4);
  _Float16* x16      = (_Float16*)wsp(d_ws, off, (size_t)R * DD * 2);     // == flat
  _Float16* wqk16    = (_Float16*)wsp(d_ws, off, (size_t)DD * 32 * 2);
  float*    bqk      = (float*)   wsp(d_ws, off, (size_t)32 * 4);
  _Float16* wv16     = (_Float16*)wsp(d_ws, off, (size_t)DD * DD * 2);
  _Float16* w1_16    = (_Float16*)wsp(d_ws, off, (size_t)256 * 256 * 2);
  _Float16* w2_16    = (_Float16*)wsp(d_ws, off, (size_t)256 * 128 * 2);
  _Float16* sw16     = (_Float16*)wsp(d_ws, off, (size_t)64 * 128 * 2);
  float*    QKf      = (float*)   wsp(d_ws, off, (size_t)R * 32 * 4);
  _Float16* Vh       = (_Float16*)wsp(d_ws, off, (size_t)R * DD * 2);
  float*    scores   = (float*)   wsp(d_ws, off, (size_t)BB * SS * SS * 4);
  _Float16* gah      = (_Float16*)wsp(d_ws, off, (size_t)BB * SS * SS * 2);
  _Float16* out2h    = (_Float16*)wsp(d_ws, off, (size_t)R * DD * 2);
  _Float16* dsa16    = (_Float16*)wsp(d_ws, off, (size_t)R * 256 * 2);
  _Float16* h1_16    = (_Float16*)wsp(d_ws, off, (size_t)R * 256 * 2);
  float*    mainf    = (float*)   wsp(d_ws, off, (size_t)R * DOUT * 4);
  float*    scf      = (float*)   wsp(d_ws, off, (size_t)R * DOUT * 4);

  // Weight conversions / packing
  qk_pack_kernel<<<(DD * 32 + 255) / 256, 256, 0, stream>>>(nsa_wq, nsa_wk, nsa_bq, nsa_bk, wqk16, bqk);
  f2h_kernel<<<(DD * DD + 255) / 256, 256, 0, stream>>>(nsa_wv, wv16, DD * DD);
  f2h_kernel<<<(256 * 256 + 255) / 256, 256, 0, stream>>>(cat_w1, w1_16, 256 * 256);
  f2h_kernel<<<(256 * 128 + 255) / 256, 256, 0, stream>>>(cat_w2, w2_16, 256 * 128);
  f2h_kernel<<<(64 * 128 + 255) / 256, 256, 0, stream>>>(sc_w, sw16, 64 * 128);

  // FPS -> centers (also writes center_xyz output), then KNN, then gather+pos MLP
  fps_kernel<<<BB, 256, 0, stream>>>(xyz, fpsstart, fps_idx, center, out);
  knn_kernel<<<(BB * SS + 255) / 256, 256, 0, stream>>>(xyz, center, knn_idx, knn_dist);
  group_pos_kernel<<<(int)((R + 255) / 256), 256, 0, stream>>>(
      xyz, points, center, knn_idx, knn_dist,
      pos_w1, pos_b1, pos_bn1, pos_w2, pos_b2, pos_bn2, x16);

  const int MB = (int)(R / 128);                  // 1024 M-tiles of 128 rows
  // QK = x@[wq|wk] + [bq|bk]  -> [R,32] f32   (BM=128, BN=32 config)
  gemm_tile_kernel<8, 1, 1, 2, false><<<dim3(1, MB, 1), 256, 0, stream>>>(
      x16, wqk16, bqk, nullptr, 0, QKf, nullptr, DD, DD, 32, 32, 0, 0, 0, 0);
  // V = x@wv + bv -> [R,128] f16
  gemm_tile_kernel<4, 2, 2, 2, false><<<dim3(DD / 64, MB, 1), 256, 0, stream>>>(
      x16, wv16, nsa_bv, nullptr, 0, nullptr, Vh, DD, DD, DD, DD, 0, 0, 0, 0);

  // Neighbor SA -> dsa[:, 0:128)
  nsa_kernel<<<BB * SS, 128, 0, stream>>>(QKf, Vh, x16, nsa_gam, dsa16);

  // GAM: scores = flat @ flat^T (batched, B transposed), softmax, out2 = ga @ flat
  gemm_tile_kernel<4, 2, 2, 2, true><<<dim3(SS / 64, SS / 128, BB), 256, 0, stream>>>(
      x16, x16, nullptr, nullptr, 0, scores, nullptr,
      KK * DD, KK * DD, KK * DD, SS,
      (long long)SS * KK * DD, (long long)SS * KK * DD, (long long)SS * SS, 0);
  gam_softmax_kernel<<<BB * SS, 256, 0, stream>>>(scores, gah);
  gemm_tile_kernel<4, 2, 2, 2, false><<<dim3(KK * DD / 64, SS / 128, BB), 256, 0, stream>>>(
      gah, x16, nullptr, nullptr, 0, nullptr, out2h,
      SS, SS, KK * DD, KK * DD,
      (long long)SS * SS, (long long)SS * KK * DD, (long long)SS * KK * DD, 0);
  grp_kernel<<<(int)((R * DD + 255) / 256), 256, 0, stream>>>(out2h, x16, gam_gam, dsa16);

  // cat MLP: H1 = relu(bn(dsa@w1+b1)) [f16], main = bn(H1@w2+b2) [f32]
  gemm_tile_kernel<4, 2, 2, 2, false><<<dim3(256 / 64, MB, 1), 256, 0, stream>>>(
      dsa16, w1_16, cat_b1, cat_bn1, 256, nullptr, h1_16,
      256, 256, 256, 256, 0, 0, 0, 1);
  gemm_tile_kernel<4, 2, 2, 2, false><<<dim3(DOUT / 64, MB, 1), 256, 0, stream>>>(
      h1_16, w2_16, cat_b2, cat_bn2, DOUT, mainf, nullptr,
      256, 256, DOUT, DOUT, 0, 0, 0, 0);

  // shortcut: sc = relu(bn(grouped_pts @ sc_w + sc_b)); grouped_pts = x cols [64,128)
  gemm_tile_kernel<4, 2, 2, 2, false><<<dim3(DOUT / 64, MB, 1), 256, 0, stream>>>(
      x16 + 64, sw16, sc_b, sc_bn, DOUT, scf, nullptr,
      DIN, DD, DOUT, DOUT, 0, 0, 0, 1);

  // new_points = max_k relu(main + sc)
  final_kernel<<<(BB * SS * DOUT + 255) / 256, 256, 0, stream>>>(mainf, scf, out);
}
